// ConvLSTMResBlock_34351148434213
// MI455X (gfx1250) — matrix operand order repaired and checked by reading, not compiled, not run on hardware
//
#include <hip/hip_runtime.h>
#include <hip/hip_bf16.h>
#include <math.h>

// ---- problem constants (match reference) ----
#define B_   8
#define T_   14
#define H_   64
#define W_   64
#define C_   64
#define F_   64
#define N4F  256            // 4*F gate channels
#define KSLABS 36           // 2 parts (x,h) * 9 taps * 2 chunks of 32 channels
#define MTILE 64            // spatial positions per block
#define NBLOCKS 16          // 256 / 16

typedef __attribute__((ext_vector_type(16))) __bf16 v16bf;
typedef __attribute__((ext_vector_type(8)))  float  v8f;

union AFrag { v16bf v; unsigned short u[16]; };

__device__ __forceinline__ unsigned short f2bf(float f) {
  // f32 -> bf16 round-to-nearest-even
  unsigned u = __float_as_uint(f);
  unsigned r = ((u >> 16) & 1u) + 0x7FFFu;
  return (unsigned short)((u + r) >> 16);
}

// ---------------------------------------------------------------------------
// Pack Wx [3,3,64,256] and Wh [3,3,64,256] (f32) into bf16 WMMA-B fragments.
// Layout: [slab s][nblock nb][lane][16 bf16]; per fragment lane%16 == N column and
// the 16 per-lane elements enumerate K {8h..8h+7, 16+8h..16+8h+7} -> one 32B load/lane.
// ---------------------------------------------------------------------------
__global__ void pack_weights_kernel(const float* __restrict__ Wx,
                                    const float* __restrict__ Wh,
                                    unsigned short* __restrict__ packed) {
  int p = blockIdx.x * blockDim.x + threadIdx.x;        // one (s,nb,lane) chunk
  if (p >= KSLABS * NBLOCKS * 32) return;
  int lane = p & 31;
  int nb   = (p >> 5) & 15;
  int s    = p >> 9;
  int part = s / 18;         // 0 = x-conv (Wx), 1 = h-conv (Wh)
  int r    = s % 18;
  int tap  = r >> 1;         // kh*3 + kw
  int cc   = r & 1;          // channel chunk (32)
  int c    = cc * 32 + ((lane >> 4) << 3); // K base for lane half
  const float* Wsrc = part ? Wh : Wx;
  unsigned short* dst = packed + (size_t)p * 16;
#pragma unroll
  for (int e = 0; e < 16; ++e) {
    int n = nb * 16 + (lane & 15);
    dst[e] = f2bf(Wsrc[((size_t)tap * 64 + c + (e < 8 ? e : e + 8)) * 256 + n]);
  }
}

__global__ void init_state_kernel(float* __restrict__ h0,
                                  float* __restrict__ c0, int n) {
  int i = blockIdx.x * blockDim.x + threadIdx.x;
  if (i < n) { h0[i] = 0.f; c0[i] = 0.f; }
}

// A-fragment gather (im2col, SAME padding -> zeros) + f32->bf16 conversion.
// coff = cc*32 + base0 selects this lane's 16 K values within the channel dim.
__device__ __forceinline__ AFrag load_a_frag(const float* __restrict__ base,
                                             int row, int col, int coff) {
  AFrag a;
  const bool valid = ((unsigned)row < (unsigned)H_) &&
                     ((unsigned)col < (unsigned)W_);
  if (valid) {
    const float* src = base + ((size_t)(row * 64 + col) * 64 + coff);
    const float4 f0 = *reinterpret_cast<const float4*>(src + 0);
    const float4 f1 = *reinterpret_cast<const float4*>(src + 4);
    const float4 f2 = *reinterpret_cast<const float4*>(src + 16);
    const float4 f3 = *reinterpret_cast<const float4*>(src + 20);
    a.u[0]  = f2bf(f0.x); a.u[1]  = f2bf(f0.y);
    a.u[2]  = f2bf(f0.z); a.u[3]  = f2bf(f0.w);
    a.u[4]  = f2bf(f1.x); a.u[5]  = f2bf(f1.y);
    a.u[6]  = f2bf(f1.z); a.u[7]  = f2bf(f1.w);
    a.u[8]  = f2bf(f2.x); a.u[9]  = f2bf(f2.y);
    a.u[10] = f2bf(f2.z); a.u[11] = f2bf(f2.w);
    a.u[12] = f2bf(f3.x); a.u[13] = f2bf(f3.y);
    a.u[14] = f2bf(f3.z); a.u[15] = f2bf(f3.w);
  } else {
#pragma unroll
    for (int e = 0; e < 16; ++e) a.u[e] = 0;
  }
  return a;
}

// ---------------------------------------------------------------------------
// One ConvLSTM timestep, fused implicit-GEMM (WMMA bf16, f32 accum) + gates.
// Block = 256 threads (8 wave32), tile = 64 M x 256 N (full gate width).
// Wave grid: waveM in [0,4) x waveN in [0,2); wave owns 16 M x 128 N.
// K loop (36 slabs of 32) is software-pipelined: loads for slab s+1 are
// issued as one clause before the 8 WMMAs of slab s consume registers.
// ---------------------------------------------------------------------------
__global__ __launch_bounds__(256, 1)
void convlstm_step_kernel(const float* __restrict__ x,       // [B,T,H,W,C]
                          const float* __restrict__ h_in,    // [B,H,W,F]
                          float* __restrict__ h_out,         // [B,H,W,F]
                          float* __restrict__ c_st,          // [B,H,W,F]
                          const unsigned short* __restrict__ packedW,
                          const float* __restrict__ bias,    // [256]
                          float* __restrict__ y,             // [B,T,H,W,F]
                          int t) {
  __shared__ float zlds[MTILE * N4F];                        // 64 KB z tile

  const int tid   = threadIdx.x;
  const int lane  = tid & 31;
  const int wave  = tid >> 5;
  const int waveM = wave >> 1;        // 0..3
  const int waveN = wave & 1;         // 0..1

  const int m0   = blockIdx.x * MTILE;
  const int mloc = waveM * 16 + (lane & 15);   // A-fragment row (M) for lane
  const int m    = m0 + mloc;
  const int bb   = m >> 12;                    // / (H*W)
  const int rem  = m & 4095;
  const int yy   = rem >> 6;
  const int xx   = rem & 63;
  const int base0 = (lane >> 4) << 3;          // lanes 0-15: K {0..7,16..23};
                                               // lanes 16-31: K {8..15,24..31}
  const v16bf* Wv = reinterpret_cast<const v16bf*>(packedW);
  const int bcol0 = (waveN * 8) * 32 + lane;   // B fragment column base

  // per-part A base pointers (batch + timestep folded in)
  const float* abase[2];
  abase[0] = x + (size_t)(bb * T_ + t) * 4096 * 64;
  abase[1] = h_in + (size_t)bb * 4096 * 64;

  v8f acc[8] = {};                             // 16x16 f32 accumulators

  // ---- pipeline prologue: slab 0 = (part 0, kh 0, kw 0, cc 0) ----
  AFrag aCur = load_a_frag(abase[0], yy - 1, xx - 1, base0);
  v16bf bCur[8];
#pragma unroll
  for (int j = 0; j < 8; ++j) bCur[j] = Wv[bcol0 + j * 32];

  // incremental slab indices (no div/mod): cc -> kw -> kh -> part
  int part = 0, kh = 0, kw = 0, cc = 0;

#pragma unroll 2
  for (int s = 0; s < KSLABS; ++s) {
    // -- next slab indices (carry chain), clamped at the tail (branch-free) --
    int ncc = cc + 1, nkw = kw, nkh = kh, npart = part;
    if (ncc == 2) { ncc = 0; ++nkw;
      if (nkw == 3) { nkw = 0; ++nkh;
        if (nkh == 3) { nkh = 0; ++npart; } } }
    if (npart > 1) npart = 1;                     // s==35: dummy in-bounds load
    const int sn = (s + 1 < KSLABS) ? (s + 1) : s;

    // -- issue all loads for slab s+1 (A: 4xb128 + cvt, B: 16xb128 clause) --
    AFrag aNxt = load_a_frag(abase[npart], yy + nkh - 1, xx + nkw - 1,
                             ncc * 32 + base0);
    v16bf bNxt[8];
    const int sbase = sn * (NBLOCKS * 32) + bcol0;
#pragma unroll
    for (int j = 0; j < 8; ++j) bNxt[j] = Wv[sbase + j * 32];

    // -- 8 WMMAs on the resident slab s fragments (overlap next loads) --
#pragma unroll
    for (int j = 0; j < 8; ++j) {
      acc[j] = __builtin_amdgcn_wmma_f32_16x16x32_bf16(
          false, aCur.v, false, bCur[j], (short)0, acc[j], false, false);
    }

    // -- rotate buffers (elided by unroll-by-2 role swap) --
    aCur = aNxt;
#pragma unroll
    for (int j = 0; j < 8; ++j) bCur[j] = bNxt[j];
    part = npart; kh = nkh; kw = nkw; cc = ncc;
  }

  // ---- spill accumulators to LDS z tile (C/D layout: VGPR r -> M, lane -> N)
#pragma unroll
  for (int j = 0; j < 8; ++j) {
    const int nl = waveN * 128 + j * 16 + (lane & 15);
    const int mb = waveM * 16 + ((lane >> 4) << 3);
#pragma unroll
    for (int rr = 0; rr < 8; ++rr) {
      zlds[(mb + rr) * N4F + nl] = acc[j][rr];
    }
  }
  __syncthreads();

  // ---- fused gates + state update + residual: 64 pos x 64 out-ch ----
#pragma unroll 4
  for (int it = 0; it < 16; ++it) {
    const int idx = it * 256 + tid;        // 0..4095
    const int ml  = idx >> 6;              // local position
    const int co  = idx & 63;              // output channel
    const int mg  = m0 + ml;
    const int b2  = mg >> 12;
    const int sp  = mg & 4095;             // y*64 + x

    const float zi = zlds[ml * N4F + co]        + bias[co];
    const float zf = zlds[ml * N4F + 64  + co]  + bias[64 + co];
    const float zg = zlds[ml * N4F + 128 + co]  + bias[128 + co];
    const float zo = zlds[ml * N4F + 192 + co]  + bias[192 + co];

    const size_t soff = (size_t)mg * 64 + co;                          // states
    const size_t goff = ((size_t)(b2 * T_ + t) * 4096 + sp) * 64 + co; // x / y

    const float cp = c_st[soff];
    const float ig = fminf(fmaxf(0.2f * zi + 0.5f, 0.f), 1.f);
    const float fg = fminf(fmaxf(0.2f * zf + 0.5f, 0.f), 1.f);
    const float og = fminf(fmaxf(0.2f * zo + 0.5f, 0.f), 1.f);
    const float gg = tanhf(zg);
    const float cn = fg * cp + ig * gg;
    const float hn = og * tanhf(cn);

    c_st[soff]  = cn;
    h_out[soff] = hn;
    y[goff]     = hn + x[goff];            // residual skip (C == F)
  }
}

// ---------------------------------------------------------------------------
extern "C" void kernel_launch(void* const* d_in, const int* in_sizes, int n_in,
                              void* d_out, int out_size, void* d_ws,
                              size_t ws_size, hipStream_t stream) {
  (void)in_sizes; (void)n_in; (void)out_size; (void)ws_size;
  const float* x  = (const float*)d_in[0];
  const float* Wx = (const float*)d_in[1];
  const float* Wh = (const float*)d_in[2];
  const float* b  = (const float*)d_in[3];
  float* y = (float*)d_out;

  char* ws = (char*)d_ws;
  unsigned short* packed = (unsigned short*)ws;            // 576 KB used
  const size_t OFF_H0 = (size_t)1 << 20;
  const size_t SBYTES = (size_t)B_ * H_ * W_ * F_ * sizeof(float); // 8 MiB
  float* h0 = (float*)(ws + OFF_H0);
  float* h1 = (float*)(ws + OFF_H0 + SBYTES);
  float* c  = (float*)(ws + OFF_H0 + 2 * SBYTES);

  // per-launch (determinism): weight packing + state zeroing
  pack_weights_kernel<<<(KSLABS * NBLOCKS * 32 + 255) / 256, 256, 0, stream>>>(
      Wx, Wh, packed);
  const int nstate = B_ * H_ * W_ * F_;
  init_state_kernel<<<(nstate + 255) / 256, 256, 0, stream>>>(h0, c, nstate);

  // sequential recurrence; h double-buffered (neighbor reads), c in place
  for (int t = 0; t < T_; ++t) {
    const float* hin = (t & 1) ? h1 : h0;
    float*      hout = (t & 1) ? h0 : h1;
    convlstm_step_kernel<<<(B_ * H_ * W_) / MTILE, 256, 0, stream>>>(
        x, hin, hout, c, packed, b, y, t);
  }
}
